// VQVAEEncoder_73555609911455
// MI455X (gfx1250) — compile-verified, matrix operand order checked by reading, not run on hardware
//
#include <hip/hip_runtime.h>
#include <hip/hip_bf16.h>

// ---------------------------------------------------------------------------
// VQ-VAE encoder for MI455X (gfx1250): bf16 WMMA GEMMs + async-LDS staged
// WMMA distance/argmin.
// ---------------------------------------------------------------------------

typedef __attribute__((ext_vector_type(16))) __bf16 v16bf;
typedef __attribute__((ext_vector_type(8)))  float  v8f;

#define NROWS 32768
#define IN_DIM 768
#define H1DIM 512
#define H2DIM 256
#define CDIM  128
#define CSIZE 8192

// LDS codebook tile: TILE_E entries x 128 bf16 (256B) + 16B row pad -> 68-word
// row stride => 16-lane b128 fragment reads are bank-conflict free.
#define TILE_E 64
#define TILE_ROW_BYTES 272
#define TILE_BYTES (TILE_E * TILE_ROW_BYTES)
#define NTILES (CSIZE / TILE_E)

#if defined(__gfx1250__) &&                                            \
    __has_builtin(__builtin_amdgcn_global_load_async_to_lds_b128) &&   \
    __has_builtin(__builtin_amdgcn_s_wait_asynccnt)
#define USE_ASYNC_COPY 1
#else
#define USE_ASYNC_COPY 0
#endif

#if USE_ASYNC_COPY
typedef int b128_t __attribute__((vector_size(16)));
typedef __attribute__((address_space(1))) b128_t* gptr_b128;
typedef __attribute__((address_space(3))) b128_t* lptr_b128;
#endif

union FragU {
    v16bf v;
    __bf16 h[16];
    unsigned u[8];
};

// A/B fragment (16-bit, 16x32 / 32x16) per ISA 7.12.2:
//   lane&15 selects row (A: M, B: N); half = lane>>4 shifts K by +8 (+24).
//   VGPR v holds K-pair base_v in {0,2,4,6,16,18,20,22} (+8*half).
// Row-major source with K contiguous -> pairs are contiguous dwords.
__device__ __forceinline__ v16bf load_frag_bf16(const __bf16* __restrict__ base,
                                                int row, int ldk, int k0, int lane) {
    const int half = lane >> 4;
    const unsigned* p = reinterpret_cast<const unsigned*>(
        base + (size_t)row * ldk + k0 + 8 * half);
    FragU f;
#pragma unroll
    for (int v = 0; v < 8; ++v)
        f.u[v] = p[(v < 4) ? v : (v + 4)];   // dwords 0..3 and 8..11
    return f.v;
}

__device__ __forceinline__ v16bf load_frag_f32(const float* __restrict__ base,
                                               int row, int ldk, int k0, int lane) {
    const int half = lane >> 4;
    const float* p = base + (size_t)row * ldk + k0 + 8 * half;
    FragU f;
#pragma unroll
    for (int v = 0; v < 8; ++v) {
        const int o = (v < 4) ? (2 * v) : (16 + 2 * (v - 4));
        f.h[2 * v + 0] = (__bf16)p[o + 0];
        f.h[2 * v + 1] = (__bf16)p[o + 1];
    }
    return f.v;
}

__device__ __forceinline__ v16bf load_frag_lds(const unsigned char* tile,
                                               int nloc, int k0, int half) {
    const unsigned* p = reinterpret_cast<const unsigned*>(
        tile + nloc * TILE_ROW_BYTES + (k0 + 8 * half) * 2);
    FragU f;
#pragma unroll
    for (int v = 0; v < 8; ++v)
        f.u[v] = p[(v < 4) ? v : (v + 4)];
    return f.v;
}

// Stage one TILE_E-entry codebook tile (16 KB) into LDS:
// 256 threads x 4 chunks x 16B (async DMA, ASYNCcnt-tracked).
__device__ __forceinline__ void stage_tile(const __bf16* __restrict__ CB, int t,
                                           unsigned char* dst, int tid) {
#pragma unroll
    for (int i = 0; i < (TILE_E * 16) / 256; ++i) {
        const int c = tid + 256 * i;
        const int e = c >> 4;    // codebook entry within tile
        const int ch = c & 15;   // 16-byte chunk within entry row
        const __bf16* src = CB + (size_t)(t * TILE_E + e) * CDIM + ch * 8;
        unsigned char* d = dst + e * TILE_ROW_BYTES + ch * 16;
#if USE_ASYNC_COPY
        __builtin_amdgcn_global_load_async_to_lds_b128(
            (gptr_b128)(void*)(src), (lptr_b128)(void*)(d), 0, 0);
#else
        *(uint4*)d = *(const uint4*)src;
#endif
    }
}

__device__ __forceinline__ void wait_stage() {
#if USE_ASYNC_COPY
    __builtin_amdgcn_s_wait_asynccnt(0);
#endif
    __syncthreads();
}

// ---------------------------------------------------------------------------
// Generic WMMA GEMM: out[M,N] = act(A[M,K] * Bt[N,K]^T + bias[N]) in bf16.
// One wave per 16x16 tile, 8 waves per block, 2-stage software pipeline on K.
// ---------------------------------------------------------------------------
template <bool RELU, bool AF32>
__global__ __launch_bounds__(256) void gemm_wmma_bf16(
    const void* __restrict__ Aptr, const __bf16* __restrict__ Bt,
    const float* __restrict__ bias, __bf16* __restrict__ out,
    int M, int K, int N) {
    const int lane = threadIdx.x & 31;
    const int wave = threadIdx.x >> 5;
    const int tiles_n = N >> 4;
    const int tile = blockIdx.x * 8 + wave;
    const int tm = tile / tiles_n;
    const int tn = tile - tm * tiles_n;
    if (tm * 16 >= M) return;
    const int m0 = tm * 16, n0 = tn * 16;
    const int arow = m0 + (lane & 15);
    const int brow = n0 + (lane & 15);

    v16bf a = AF32 ? load_frag_f32((const float*)Aptr, arow, K, 0, lane)
                   : load_frag_bf16((const __bf16*)Aptr, arow, K, 0, lane);
    v16bf b = load_frag_bf16(Bt, brow, K, 0, lane);
    v8f acc = {};
    for (int k0 = 32; k0 < K; k0 += 32) {
        // issue next-iteration loads before the WMMA's loadcnt wait
        v16bf an = AF32 ? load_frag_f32((const float*)Aptr, arow, K, k0, lane)
                        : load_frag_bf16((const __bf16*)Aptr, arow, K, k0, lane);
        v16bf bn = load_frag_bf16(Bt, brow, K, k0, lane);
        acc = __builtin_amdgcn_wmma_f32_16x16x32_bf16(
            false, a, false, b, (short)0, acc, false, false);
        a = an;
        b = bn;
    }
    acc = __builtin_amdgcn_wmma_f32_16x16x32_bf16(
        false, a, false, b, (short)0, acc, false, false);

    const int n = n0 + (lane & 15);
    const int half = lane >> 4;
    const float bv = bias[n];
#pragma unroll
    for (int r = 0; r < 8; ++r) {
        float v = acc[r] + bv;
        if (RELU) v = v > 0.0f ? v : 0.0f;
        out[(size_t)(m0 + r + 8 * half) * N + n] = (__bf16)v;
    }
}

// ---------------------------------------------------------------------------
// Distance + argmin. Block = 8 waves = 128 z-rows. All waves share 64-entry
// codebook tiles staged into LDS via double-buffered async global->LDS DMA
// (16 WMMAs per wait+barrier); each wave keeps its 16x128 z-tile in registers
// (4 A-fragments). argmin over d2' = ||c||^2 - 2 z.c (||z||^2 dropped:
// row-constant, argmin invariant). Tie-break toward lower index (jnp.argmin).
// ---------------------------------------------------------------------------
__global__ __launch_bounds__(256) void vq_argmin_wmma(
    const __bf16* __restrict__ Z, const __bf16* __restrict__ CB,
    const float* __restrict__ c2, int* __restrict__ codes) {
    __shared__ __align__(16) unsigned char sh[2][TILE_BYTES];
    const int tid = threadIdx.x;
    const int lane = tid & 31;
    const int wave = tid >> 5;
    const int m0 = (blockIdx.x * 8 + wave) * 16;
    const int half = lane >> 4;
    const int nloc = lane & 15;
    const int arow = m0 + nloc;

    const v16bf a0 = load_frag_bf16(Z, arow, CDIM, 0, lane);
    const v16bf a1 = load_frag_bf16(Z, arow, CDIM, 32, lane);
    const v16bf a2 = load_frag_bf16(Z, arow, CDIM, 64, lane);
    const v16bf a3 = load_frag_bf16(Z, arow, CDIM, 96, lane);

    float bestd[8];
    int besti[8];
#pragma unroll
    for (int r = 0; r < 8; ++r) { bestd[r] = 3.4e38f; besti[r] = 0; }

    stage_tile(CB, 0, sh[0], tid);
    wait_stage();

    int buf = 0;
    for (int t = 0; t < NTILES; ++t) {
        if (t + 1 < NTILES) stage_tile(CB, t + 1, sh[buf ^ 1], tid);

        const unsigned char* tile = sh[buf];
#pragma unroll
        for (int sub = 0; sub < TILE_E / 16; ++sub) {
            const unsigned char* st = tile + sub * 16 * TILE_ROW_BYTES;
            v16bf b0 = load_frag_lds(st, nloc, 0, half);
            v16bf b1 = load_frag_lds(st, nloc, 32, half);
            v16bf b2 = load_frag_lds(st, nloc, 64, half);
            v16bf b3 = load_frag_lds(st, nloc, 96, half);
            v8f acc = {};
            acc = __builtin_amdgcn_wmma_f32_16x16x32_bf16(false, a0, false, b0, (short)0, acc, false, false);
            acc = __builtin_amdgcn_wmma_f32_16x16x32_bf16(false, a1, false, b1, (short)0, acc, false, false);
            acc = __builtin_amdgcn_wmma_f32_16x16x32_bf16(false, a2, false, b2, (short)0, acc, false, false);
            acc = __builtin_amdgcn_wmma_f32_16x16x32_bf16(false, a3, false, b3, (short)0, acc, false, false);

            const int n = t * TILE_E + sub * 16 + nloc;
            const float cc = c2[n];
#pragma unroll
            for (int r = 0; r < 8; ++r) {
                const float d = cc - 2.0f * acc[r];
                if (d < bestd[r]) { bestd[r] = d; besti[r] = n; }
            }
        }

        wait_stage();   // tile t+1 resident; all waves done reading sh[buf]
        buf ^= 1;
    }

#pragma unroll
    for (int r = 0; r < 8; ++r) {
        float d = bestd[r];
        int i = besti[r];
#pragma unroll
        for (int off = 1; off < 16; off <<= 1) {
            const float od = __shfl_xor(d, off, 32);
            const int oi = __shfl_xor(i, off, 32);
            if (od < d || (od == d && oi < i)) { d = od; i = oi; }
        }
        if (nloc == 0) codes[m0 + r + 8 * half] = i;
    }
}

// ---------------------------------------------------------------------------
// Prep / epilogue kernels
// ---------------------------------------------------------------------------
__global__ void k_transpose_bf16(const float* __restrict__ W, __bf16* __restrict__ Wt,
                                 int K, int N) {  // W[K,N] -> Wt[N,K] bf16
    const int t = blockIdx.x * blockDim.x + threadIdx.x;
    if (t >= K * N) return;
    const int n = t / K;
    const int k = t - n * K;
    Wt[t] = (__bf16)W[(size_t)k * N + n];
}

__global__ void k_convert_bf16(const float* __restrict__ src, __bf16* __restrict__ dst,
                               int total) {
    const int t = blockIdx.x * blockDim.x + threadIdx.x;
    if (t < total) dst[t] = (__bf16)src[t];
}

__global__ void k_codebook_norm2(const float* __restrict__ cb, float* __restrict__ c2) {
    const int c = blockIdx.x * blockDim.x + threadIdx.x;
    if (c >= CSIZE) return;
    const float* p = cb + (size_t)c * CDIM;
    float s = 0.0f;
#pragma unroll 8
    for (int d = 0; d < CDIM; ++d) s += p[d] * p[d];
    c2[c] = s;
}

__global__ void k_gather(const float* __restrict__ cb, const int* __restrict__ codes,
                         float* __restrict__ outq, int total) {
    const int t = blockIdx.x * blockDim.x + threadIdx.x;
    if (t >= total) return;
    const int n = t >> 7;        // /CDIM
    const int d = t & (CDIM - 1);
    outq[t] = cb[(size_t)codes[n] * CDIM + d];
}

// ---------------------------------------------------------------------------
extern "C" void kernel_launch(void* const* d_in, const int* in_sizes, int n_in,
                              void* d_out, int out_size, void* d_ws, size_t ws_size,
                              hipStream_t stream) {
    (void)in_sizes; (void)n_in; (void)out_size; (void)ws_size;
    const float* x  = (const float*)d_in[0];
    const float* W1 = (const float*)d_in[1];
    const float* b1 = (const float*)d_in[2];
    const float* W2 = (const float*)d_in[3];
    const float* b2 = (const float*)d_in[4];
    const float* W3 = (const float*)d_in[5];
    const float* b3 = (const float*)d_in[6];
    const float* cb = (const float*)d_in[7];

    float* quant = (float*)d_out;                         // [NROWS, CDIM] f32
    int*   codes = (int*)(quant + (size_t)NROWS * CDIM);  // [NROWS] i32

    auto alignup = [](size_t v) { return (v + 255) & ~(size_t)255; };
    char* w = (char*)d_ws;
    __bf16* W1t = (__bf16*)w; w += alignup((size_t)H1DIM * IN_DIM * 2);
    __bf16* W2t = (__bf16*)w; w += alignup((size_t)H2DIM * H1DIM * 2);
    __bf16* W3t = (__bf16*)w; w += alignup((size_t)CDIM * H2DIM * 2);
    __bf16* cbb = (__bf16*)w; w += alignup((size_t)CSIZE * CDIM * 2);
    float*  c2  = (float*)w;  w += alignup((size_t)CSIZE * 4);
    __bf16* h1  = (__bf16*)w; w += alignup((size_t)NROWS * H1DIM * 2);
    __bf16* h2  = (__bf16*)w; w += alignup((size_t)NROWS * H2DIM * 2);
    __bf16* zb  = (__bf16*)w; w += alignup((size_t)NROWS * CDIM * 2);

    // --- weight / codebook prep ---
    k_transpose_bf16<<<(IN_DIM * H1DIM + 255) / 256, 256, 0, stream>>>(W1, W1t, IN_DIM, H1DIM);
    k_transpose_bf16<<<(H1DIM * H2DIM + 255) / 256, 256, 0, stream>>>(W2, W2t, H1DIM, H2DIM);
    k_transpose_bf16<<<(H2DIM * CDIM + 255) / 256, 256, 0, stream>>>(W3, W3t, H2DIM, CDIM);
    k_convert_bf16<<<(CSIZE * CDIM + 255) / 256, 256, 0, stream>>>(cb, cbb, CSIZE * CDIM);
    k_codebook_norm2<<<(CSIZE + 255) / 256, 256, 0, stream>>>(cb, c2);

    // --- MLP encoder ---
    {
        const int tiles = (NROWS / 16) * (H1DIM / 16);
        gemm_wmma_bf16<true, true><<<tiles / 8, 256, 0, stream>>>(
            (const void*)x, W1t, b1, h1, NROWS, IN_DIM, H1DIM);
    }
    {
        const int tiles = (NROWS / 16) * (H2DIM / 16);
        gemm_wmma_bf16<true, false><<<tiles / 8, 256, 0, stream>>>(
            (const void*)h1, W2t, b2, h2, NROWS, H1DIM, H2DIM);
    }
    {
        const int tiles = (NROWS / 16) * (CDIM / 16);
        gemm_wmma_bf16<false, false><<<tiles / 8, 256, 0, stream>>>(
            (const void*)h2, W3t, b3, zb, NROWS, H2DIM, CDIM);
    }

    // --- nearest codebook entry (async-LDS staged codebook) ---
    vq_argmin_wmma<<<NROWS / 128, 256, 0, stream>>>(zb, cbb, c2, codes);

    // --- gather quantized vectors (f32 from original codebook) ---
    k_gather<<<(NROWS * CDIM) / 256, 256, 0, stream>>>(cb, codes, quant, NROWS * CDIM);
}
